// WormholeAttention_56762287784298
// MI455X (gfx1250) — matrix operand order repaired and checked by reading, not compiled, not run on hardware
//
#include <hip/hip_runtime.h>
#include <hip/hip_bf16.h>

typedef __attribute__((ext_vector_type(16))) __bf16 bf16x16;
typedef __attribute__((ext_vector_type(8)))  __bf16 bf16x8;
typedef __attribute__((ext_vector_type(8)))  float  f32x8;

// ---------------------------------------------------------------------------
// Fragment helpers (wave32; A/B use this code's contiguous k order, C/D is the hardware layout)
//  A (16x32): lane holds row M=lane%16; elems 0..15 -> K = laneHalf*16 + e
//  B (32x16): lane holds col N=lane%16; elems 0..15 -> K = laneHalf*16 + e
//  C/D (16x16 f32): lane holds N=lane%16; vgpr r -> M = r + 8*laneHalf
// ---------------------------------------------------------------------------
__device__ __forceinline__ bf16x16 a_frag_ld(const __bf16* rowp, int lh) {
  bf16x8 lo = *(const bf16x8*)(rowp + lh * 16);
  bf16x8 hi = *(const bf16x8*)(rowp + lh * 16 + 8);
  return __builtin_shufflevector(lo, hi, 0, 1, 2, 3, 4, 5, 6, 7,
                                 8, 9, 10, 11, 12, 13, 14, 15);
}

__device__ __forceinline__ f32x8 wmma_bf16(bf16x16 a, bf16x16 b, f32x8 c) {
  return __builtin_amdgcn_wmma_f32_16x16x32_bf16(false, a, false, b,
                                                 (short)0, c, false, false);
}

// Raw LDS byte offset of a generic LDS pointer (low 32 bits of generic addr).
__device__ __forceinline__ unsigned lds_off32(const void* p) {
  return (unsigned)(uintptr_t)p;
}

// ---------------------------------------------------------------------------
// Generic GEMM:  C[M,N] = A[M,K] * W[N,K]^T + bias[N]   (f32 in, bf16 WMMA)
// Output type is a compile-time parameter (float or __bf16) so the epilogue
// is branch-free.
// block = 128 threads (4 waves), tile 64x64, K step 32.
// Requires: M%64==0, N%64==0, K%32==0 (true for every call site here).
// ---------------------------------------------------------------------------
template <typename OutT>
__global__ __launch_bounds__(128) void gemm_bias_kernel(
    const float* __restrict__ A, const float* __restrict__ W,
    const float* __restrict__ bias, OutT* __restrict__ Cout,
    int M, int N, int K) {
  __shared__ __bf16 ldsA[64 * 32];
  __shared__ __bf16 ldsB[64 * 32];
  const int t = threadIdx.x;
  const int w = t >> 5;
  const int lane = t & 31;
  const int l16 = lane & 15;
  const int lh = lane >> 4;
  const int m0 = blockIdx.y * 64;
  const int n0 = blockIdx.x * 64;
  const int mw = (w & 1) * 32;
  const int nw = (w >> 1) * 32;

  f32x8 acc[2][2];
#pragma unroll
  for (int mi = 0; mi < 2; ++mi)
#pragma unroll
    for (int ni = 0; ni < 2; ++ni)
#pragma unroll
      for (int e = 0; e < 8; ++e) acc[mi][ni][e] = 0.0f;

  const int r = t >> 1;            // 0..63 : tile row
  const int kb = (t & 1) * 16;     // 0/16  : 16 consecutive k's per thread

  for (int k0 = 0; k0 < K; k0 += 32) {
    __syncthreads();
    const float* ap = A + (size_t)(m0 + r) * K + k0 + kb;
    const float* wp = W + (size_t)(n0 + r) * K + k0 + kb;
    if (k0 + 32 < K) {
      __builtin_prefetch((const void*)(ap + 32), 0, 0);
      __builtin_prefetch((const void*)(wp + 32), 0, 0);
    }
#pragma unroll
    for (int u = 0; u < 4; ++u) {
      float4 av = *(const float4*)(ap + u * 4);
      __bf16* da = ldsA + r * 32 + kb + u * 4;
      da[0] = (__bf16)av.x; da[1] = (__bf16)av.y;
      da[2] = (__bf16)av.z; da[3] = (__bf16)av.w;
      float4 wv = *(const float4*)(wp + u * 4);
      __bf16* dw = ldsB + r * 32 + kb + u * 4;
      dw[0] = (__bf16)wv.x; dw[1] = (__bf16)wv.y;
      dw[2] = (__bf16)wv.z; dw[3] = (__bf16)wv.w;
    }
    __syncthreads();
#pragma unroll
    for (int mi = 0; mi < 2; ++mi) {
      bf16x16 af = a_frag_ld(ldsA + (mw + mi * 16 + l16) * 32, lh);
#pragma unroll
      for (int ni = 0; ni < 2; ++ni) {
        bf16x16 bfm = *(const bf16x16*)(ldsB + (nw + ni * 16 + l16) * 32 + lh * 16);
        acc[mi][ni] = wmma_bf16(af, bfm, acc[mi][ni]);
      }
    }
  }

#pragma unroll
  for (int mi = 0; mi < 2; ++mi)
#pragma unroll
    for (int ni = 0; ni < 2; ++ni) {
      const int col = n0 + nw + ni * 16 + l16;
      const float bv = bias[col];
#pragma unroll
      for (int rr = 0; rr < 8; ++rr) {
        const int row = m0 + mw + mi * 16 + rr + 8 * lh;
        Cout[(size_t)row * N + col] = (OutT)(acc[mi][ni][rr] + bv);
      }
    }
}

// ---------------------------------------------------------------------------
// Flash attention for one (i,j) pair (bf16 q/k/v, f32 output):
//   co[b,s,:] = softmax(q_i[b,s,:] . k_j[b,:,:]^T / sqrt(128)) @ v_j[b,:,:]
// q: [B,S,128] bf16, k: [B,S,128] bf16, v: [B,S,256] bf16, co: [B,S,256] f32
// block = 128 threads (4 waves); wave owns 16 query rows; 32-key steps.
// K tile is staged global->LDS with the async-to-LDS engine (ASYNCcnt).
// ---------------------------------------------------------------------------
#define AT_S 2048

__global__ __launch_bounds__(128) void attn_kernel(
    const __bf16* __restrict__ q, const __bf16* __restrict__ k,
    const __bf16* __restrict__ v, float* __restrict__ co) {
  __shared__ __bf16 kbuf[32 * 128];     // [key][dim]   (raw async copy)
  __shared__ __bf16 vbufT[256 * 32];    // [col][key]   (transposed stage)
  __shared__ __bf16 pbuf[4 * 16 * 32];  // per-wave 16x32 P tile

  const int t = threadIdx.x;
  const int w = t >> 5;
  const int lane = t & 31;
  const int l16 = lane & 15;
  const int lh = lane >> 4;
  const int bIdx = blockIdx.y;
  const int qbase = blockIdx.x * 64 + w * 16;

  const __bf16* qb = q + (size_t)bIdx * AT_S * 128;
  const __bf16* kp = k + (size_t)bIdx * AT_S * 128;
  const __bf16* vp = v + (size_t)bIdx * AT_S * 256;

  // Q fragments: 4 chunks of K (128 = 4*32), A-layout, direct bf16 loads.
  bf16x16 qf[4];
  {
    const __bf16* qrow = qb + (size_t)(qbase + l16) * 128;
#pragma unroll
    for (int c = 0; c < 4; ++c) qf[c] = a_frag_ld(qrow + c * 32, lh);
  }

  float mrun[8], lrun[8];
#pragma unroll
  for (int r2 = 0; r2 < 8; ++r2) { mrun[r2] = -3.0e38f; lrun[r2] = 0.0f; }
  f32x8 o[16];
#pragma unroll
  for (int nt = 0; nt < 16; ++nt)
#pragma unroll
    for (int e = 0; e < 8; ++e) o[nt][e] = 0.0f;

  const float scale = 0.0883883476483184f;  // 1/sqrt(128)

  for (int kb0 = 0; kb0 < AT_S; kb0 += 32) {
    __syncthreads();
    // ---- stage K tile [32,128] bf16: raw async copy global -> LDS.
    // 8 KB = 512 x b128; each of 128 lanes issues 4 transfers.
    {
#pragma unroll
      for (int u = 0; u < 4; ++u) {
        const unsigned chunk = (unsigned)t + 128u * u;          // 16B chunks
        const unsigned ldst = lds_off32(kbuf) + chunk * 16u;
        const __bf16* gsrc = kp + (size_t)kb0 * 128 + (size_t)chunk * 8;
        asm volatile("global_load_async_to_lds_b128 %0, %1, off"
                     :: "v"(ldst), "v"(gsrc) : "memory");
      }
      // ---- stage V tile transposed [256,32] (bf16 copy + transpose)
      const int key = t >> 2;
      const int cb = (t & 3) * 64;
      const __bf16* vsp = vp + (size_t)(kb0 + key) * 256 + cb;
      if (kb0 + 32 < AT_S) __builtin_prefetch((const void*)(vsp + 32 * 256), 0, 0);
#pragma unroll
      for (int u = 0; u < 8; ++u) {
        bf16x8 x = *(const bf16x8*)(vsp + u * 8);
#pragma unroll
        for (int h = 0; h < 8; ++h)
          vbufT[(cb + u * 8 + h) * 32 + key] = x[h];
      }
      asm volatile("s_wait_asynccnt 0" ::: "memory");
    }
    __syncthreads();

    // ---- scores: two 16-key subtiles, each sum of 4 K-chunk WMMAs
    f32x8 s[2];
#pragma unroll
    for (int st = 0; st < 2; ++st) {
      f32x8 acc = {};
      const __bf16* kr = kbuf + (st * 16 + l16) * 128 + lh * 16;
#pragma unroll
      for (int c = 0; c < 4; ++c) {
        bf16x16 bk = *(const bf16x16*)(kr + c * 32);
        acc = wmma_bf16(qf[c], bk, acc);
      }
#pragma unroll
      for (int e = 0; e < 8; ++e) acc[e] *= scale;
      s[st] = acc;
    }

    // ---- online softmax (row stats live in lane-halves; reduce over 16 lanes)
    float corr[8];
#pragma unroll
    for (int r2 = 0; r2 < 8; ++r2) {
      float mx = fmaxf(s[0][r2], s[1][r2]);
#pragma unroll
      for (int msk = 1; msk < 16; msk <<= 1) mx = fmaxf(mx, __shfl_xor(mx, msk, 32));
      const float mn = fmaxf(mrun[r2], mx);
      const float c0 = __expf(mrun[r2] - mn);
      const float p0 = __expf(s[0][r2] - mn);
      const float p1 = __expf(s[1][r2] - mn);
      s[0][r2] = p0; s[1][r2] = p1;
      float rs = p0 + p1;
#pragma unroll
      for (int msk = 1; msk < 16; msk <<= 1) rs += __shfl_xor(rs, msk, 32);
      lrun[r2] = lrun[r2] * c0 + rs;
      mrun[r2] = mn;
      corr[r2] = c0;
    }
#pragma unroll
    for (int nt = 0; nt < 16; ++nt)
#pragma unroll
      for (int r2 = 0; r2 < 8; ++r2) o[nt][r2] *= corr[r2];

    // ---- transpose P (C-layout -> A-layout) through per-wave LDS tile
    __bf16* pw_ = pbuf + w * 512;
#pragma unroll
    for (int st = 0; st < 2; ++st)
#pragma unroll
      for (int r2 = 0; r2 < 8; ++r2)
        pw_[(r2 + 8 * lh) * 32 + st * 16 + l16] = (__bf16)s[st][r2];
    asm volatile("s_wait_dscnt 0" ::: "memory");
    bf16x16 pf = a_frag_ld(pw_ + l16 * 32, lh);

    // ---- O += P @ V  (16 N-tiles of 16)
#pragma unroll
    for (int nt = 0; nt < 16; ++nt) {
      bf16x16 bv = *(const bf16x16*)(vbufT + (nt * 16 + l16) * 32 + lh * 16);
      o[nt] = wmma_bf16(pf, bv, o[nt]);
    }
  }

  // ---- normalize and store
  float inv[8];
#pragma unroll
  for (int r2 = 0; r2 < 8; ++r2) inv[r2] = 1.0f / lrun[r2];
#pragma unroll
  for (int nt = 0; nt < 16; ++nt)
#pragma unroll
    for (int r2 = 0; r2 < 8; ++r2) {
      const size_t row = (size_t)bIdx * AT_S + qbase + r2 + 8 * lh;
      co[row * 256 + nt * 16 + l16] = o[nt][r2] * inv[r2];
    }
}

// ---------------------------------------------------------------------------
// Elementwise helpers
// ---------------------------------------------------------------------------
__global__ __launch_bounds__(256) void concat_kernel(
    const float* __restrict__ a, const float* __restrict__ b,
    float* __restrict__ c, int d, size_t n) {
  size_t idx = (size_t)blockIdx.x * 256 + threadIdx.x;
  if (idx >= n) return;
  const int d2 = 2 * d;
  size_t row = idx / d2;
  int col = (int)(idx - row * d2);
  c[idx] = (col < d) ? a[row * d + col] : b[row * d + (col - d)];
}

__global__ __launch_bounds__(256) void gate_accum_kernel(
    const float* __restrict__ glog, const float* __restrict__ proj,
    float* __restrict__ cross, size_t n, int first) {
  size_t idx = (size_t)blockIdx.x * 256 + threadIdx.x;
  if (idx >= n) return;
  const float x = glog[idx];
  const float g = 1.0f / (1.0f + __expf(-x));
  const float val = (x > 0.0f) ? g * proj[idx] : 0.0f;  // sigmoid>0.5 <=> x>0
  cross[idx] = (first ? 0.0f : cross[idx]) + val;
}

__global__ __launch_bounds__(256) void finalize_kernel(
    const float* __restrict__ band, const float* __restrict__ cross,
    float* __restrict__ out, size_t n) {
  size_t idx = (size_t)blockIdx.x * 256 + threadIdx.x;
  if (idx >= n) return;
  out[idx] = band[idx] + 0.5f * cross[idx];
}

// ---------------------------------------------------------------------------
// Host orchestration
// ---------------------------------------------------------------------------
static inline void launch_gemm_f32(const float* A, const float* W, const float* b,
                                   float* C, int M, int N, int K, hipStream_t s) {
  dim3 grid(N / 64, M / 64);
  gemm_bias_kernel<float><<<grid, 128, 0, s>>>(A, W, b, C, M, N, K);
}

static inline void launch_gemm_bf16(const float* A, const float* W, const float* b,
                                    __bf16* C, int M, int N, int K, hipStream_t s) {
  dim3 grid(N / 64, M / 64);
  gemm_bias_kernel<__bf16><<<grid, 128, 0, s>>>(A, W, b, C, M, N, K);
}

extern "C" void kernel_launch(void* const* d_in, const int* in_sizes, int n_in,
                              void* d_out, int out_size, void* d_ws, size_t ws_size,
                              hipStream_t stream) {
  (void)in_sizes; (void)n_in; (void)out_size; (void)ws_size;
  const int dsz[3] = {256, 512, 1024};
  const int M = 4 * 2048;  // B*S

  const float* band[3]; const float* qw[3]; const float* qbv[3];
  const float* kw[3]; const float* kbv[3]; const float* vw[3]; const float* vbv[3];
  const float* pw[3]; const float* pbv[3]; const float* gw[3]; const float* gbv[3];
  for (int b = 0; b < 3; ++b) {
    band[b] = (const float*)d_in[b];
    qw[b]  = (const float*)d_in[3 + b];   qbv[b] = (const float*)d_in[6 + b];
    kw[b]  = (const float*)d_in[9 + b];   kbv[b] = (const float*)d_in[12 + b];
    vw[b]  = (const float*)d_in[15 + b];  vbv[b] = (const float*)d_in[18 + b];
    pw[b]  = (const float*)d_in[21 + b];  pbv[b] = (const float*)d_in[24 + b];
    gw[b]  = (const float*)d_in[27 + b];  gbv[b] = (const float*)d_in[30 + b];
  }

  float* ws = (float*)d_ws;
  size_t off = 0;
  // q/k/v stored as bf16 (regions sized in float units; half is slack)
  __bf16* qs[3]; __bf16* ks[3]; __bf16* vs[3];
  for (int b = 0; b < 3; ++b) { qs[b] = (__bf16*)(ws + off); off += (size_t)M * 128; }
  for (int b = 0; b < 3; ++b) { ks[b] = (__bf16*)(ws + off); off += (size_t)M * 128; }
  for (int b = 0; b < 3; ++b) { vs[b] = (__bf16*)(ws + off); off += (size_t)M * 256; }
  float* co   = ws + off; off += (size_t)M * 256;
  float* proj = ws + off; off += (size_t)M * 1024;
  float* cat  = ws + off; off += (size_t)M * 2048;
  float* glog = ws + off; off += (size_t)M * 1024;
  float* cross[3];
  for (int b = 0; b < 3; ++b) { cross[b] = ws + off; off += (size_t)M * dsz[b]; }

  // 1) QKV projections for every band -> bf16 (k_j / v_j are i-independent).
  for (int b = 0; b < 3; ++b) {
    launch_gemm_bf16(band[b], qw[b], qbv[b], qs[b], M, 128, dsz[b], stream);
    launch_gemm_bf16(band[b], kw[b], kbv[b], ks[b], M, 128, dsz[b], stream);
    launch_gemm_bf16(band[b], vw[b], vbv[b], vs[b], M, 256, dsz[b], stream);
  }

  // 2) Per (i,j) pair: attention -> out-proj -> gate -> accumulate.
  for (int i = 0; i < 3; ++i) {
    int first = 1;
    const size_t nd  = (size_t)M * dsz[i];
    const size_t nd2 = (size_t)M * 2 * dsz[i];
    for (int j = 0; j < 3; ++j) {
      if (j == i) continue;
      attn_kernel<<<dim3(2048 / 64, 4), 128, 0, stream>>>(qs[i], ks[j], vs[j], co);
      launch_gemm_f32(co, pw[i], pbv[i], proj, M, dsz[i], 256, stream);
      concat_kernel<<<(nd2 + 255) / 256, 256, 0, stream>>>(band[i], proj, cat, dsz[i], nd2);
      launch_gemm_f32(cat, gw[i], gbv[i], glog, M, dsz[i], 2 * dsz[i], stream);
      gate_accum_kernel<<<(nd + 255) / 256, 256, 0, stream>>>(glog, proj, cross[i], nd, first);
      first = 0;
    }
  }

  // 3) Finalize: out_i = band_i + cross_i / 2   (concatenated in d_out)
  float* out = (float*)d_out;
  size_t ooff = 0;
  for (int b = 0; b < 3; ++b) {
    const size_t n = (size_t)M * dsz[b];
    finalize_kernel<<<(n + 255) / 256, 256, 0, stream>>>(band[b], cross[b], out + ooff, n);
    ooff += n;
  }
}
